// BiLSTM_CRF_51247549776540
// MI455X (gfx1250) — matrix operand order repaired and checked by reading, not compiled, not run on hardware
//
#include <hip/hip_runtime.h>
#include <hip/hip_bf16.h>

#define L_SENT 2048
#define E_DIM  512
#define H2_DIM 512
#define G_DIM  2048   // 4*H2
#define NTAG   12
#define START_TAG 10
#define STOP_TAG  11
#define NEG_INF  (-10000.0f)

typedef __attribute__((ext_vector_type(16))) __bf16       v16bf;
typedef __attribute__((ext_vector_type(8)))  float        v8f;
typedef __attribute__((ext_vector_type(4)))  unsigned int u32x4;
typedef __attribute__((ext_vector_type(8)))  int          i32x8;
typedef __attribute__((ext_vector_type(4)))  int          i32x4;

union BFFrag { u32x4 u[2]; v16bf v; };

__device__ __forceinline__ unsigned short f2bf(float f) {
    unsigned int u = __float_as_uint(f);
    u += 0x7FFFu + ((u >> 16) & 1u);   // round-to-nearest-even
    return (unsigned short)(u >> 16);
}

__device__ __forceinline__ float sigmoidf_(float x) {
    return 1.0f / (1.0f + __expf(-x));
}

// ---------------------------------------------------------------- conversions
__global__ void f32_to_bf16_kernel(const float* __restrict__ src,
                                   unsigned short* __restrict__ dst, int n) {
    int i = blockIdx.x * blockDim.x + threadIdx.x;
    if (i < n) dst[i] = f2bf(src[i]);
}

__global__ void gather_embed_kernel(const int* __restrict__ sent,
                                    const float* __restrict__ emb,
                                    unsigned short* __restrict__ xs_bf) {
    int i = blockIdx.x * blockDim.x + threadIdx.x;   // over L*E
    if (i >= L_SENT * E_DIM) return;
    int t = i >> 9;        // / 512
    int e = i & 511;
    xs_bf[i] = f2bf(emb[(size_t)sent[t] * E_DIM + e]);
}

// ------------------------------------------------- input projection (WMMA GEMM)
// xw[t, g] = sum_e xs[t, e] * W_ih[g, e] + b[g]   (dir 1 uses reversed xs rows)
// A tile (16 x 512 bf16, shared by all 8 waves of the block) is staged into LDS
// once per block by the Tensor Data Mover; B streams from L2 per wave.
__global__ __launch_bounds__(256)
void input_gemm_kernel(const unsigned short* __restrict__ xs_bf,
                       const unsigned short* __restrict__ Wf,
                       const unsigned short* __restrict__ Wb,
                       const float* __restrict__ bias_f,
                       const float* __restrict__ bias_b,
                       float* __restrict__ xw_f,
                       float* __restrict__ xw_b) {
    __shared__ unsigned short aTile[16 * E_DIM];   // 16 KB

    const int lane  = threadIdx.x & 31;
    const int wv    = threadIdx.x >> 5;
    const int row16 = lane & 15;
    const int half  = lane >> 4;
    const int mt    = blockIdx.x;                 // time tile (0..127)
    const int nt    = blockIdx.y * 8 + wv;        // gate tile (0..127)
    const int dir   = blockIdx.z;

    const unsigned short* W    = dir ? Wb     : Wf;
    const float*          bias = dir ? bias_b : bias_f;
    float*                xw   = dir ? xw_b   : xw_f;

    // ---- TDM: stage A tile into LDS (wave 0 issues the descriptor) ----
    if (wv == 0) {
        // dir==1 needs rows L-1-(mt*16+m); those are the ascending contiguous
        // range starting at r0 = L-16-mt*16 (read back flipped below).
        const int r0 = dir ? (L_SENT - 16 - mt * 16) : (mt * 16);
        unsigned long long ga =
            (unsigned long long)(const void*)(xs_bf + (size_t)r0 * E_DIM);
        unsigned ldsoff = (unsigned)(unsigned long long)(const void*)aTile;
        // D# group0: count=1 | lds_addr | global_addr[56:0] | type=2
        u32x4 g0 = { 1u,
                     ldsoff,
                     (unsigned)ga,
                     ((unsigned)((ga >> 32) & 0x01FFFFFFull)) | (2u << 30) };
        // D# group1: data_size=1(2B); tensor_dim0=512; tensor_dim1=2048;
        //            tile_dim0=512; tile_dim1=16; tensor_dim0_stride=512
        i32x8 g1 = { 0x00010000,            // workgroup_mask=0, data_size=1
                     (int)(512u  << 16),    // tensor_dim0[15:0] @ bits 63:48
                     (int)(2048u << 16),    // tensor_dim1[15:0] @ bits 95:80
                     (int)(512u  << 16),    // tile_dim0 @ bits 127:112
                     16,                    // tile_dim1 @ bits 143:128
                     512,                   // tensor_dim0_stride[31:0]
                     0, 0 };
        i32x4 gz4 = { 0, 0, 0, 0 };         // groups 2/3 unused (2D tensor)
        i32x8 gz8 = { 0, 0, 0, 0, 0, 0, 0, 0 };
        __builtin_amdgcn_tensor_load_to_lds(g0, g1, gz4, gz4, gz8, 0);
        __builtin_amdgcn_s_wait_tensorcnt(0);
    }
    __syncthreads();

    const int aRowLds = dir ? (15 - row16) : row16;  // un-flip reversed rows
    const unsigned short* aBase = &aTile[aRowLds * E_DIM];
    const int g = nt * 16 + row16;
    const unsigned short* bBase = W + (size_t)g * E_DIM;

    v8f acc = {};
    for (int k0 = 0; k0 < E_DIM; k0 += 32) {
        BFFrag a, b;
        // A 16x32 bf16 layout: lane holds K = {8h..8h+7} and {16+8h..16+8h+7}
        a.u[0] = *(const u32x4*)(aBase + k0 + 8 * half);
        a.u[1] = *(const u32x4*)(aBase + k0 + 16 + 8 * half);
        // B 32x16 bf16 layout: lane holds column g, K = {8h..8h+7} and {16+8h..16+8h+7}
        b.u[0] = *(const u32x4*)(bBase + k0 + 8 * half);
        b.u[1] = *(const u32x4*)(bBase + k0 + 16 + 8 * half);
        acc = __builtin_amdgcn_wmma_f32_16x16x32_bf16(
                  false, a.v, false, b.v, (short)0, acc, false, false);
    }
    const float bv = bias[nt * 16 + row16];       // bias per column N = lane&15
#pragma unroll
    for (int r = 0; r < 8; ++r) {
        const int m = mt * 16 + r + 8 * half;     // D row = r + 8*(lane>>4)
        xw[(size_t)m * G_DIM + nt * 16 + row16] = acc[r] + bv;
    }
}

// ------------------------------------------------------ LSTM recurrence (WMMA)
// One workgroup per direction; per step: gw = h @ W_hh^T via WMMA with h
// broadcast into every A row, then gate nonlinearity + state update.
__global__ __launch_bounds__(1024)
void lstm_recurrence_kernel(const float* __restrict__ xw_f,
                            const float* __restrict__ xw_b,
                            const unsigned short* __restrict__ Whh_f_bf,
                            const unsigned short* __restrict__ Whh_b_bf,
                            const float* __restrict__ h0,
                            const float* __restrict__ c0,
                            float* __restrict__ hf,
                            float* __restrict__ hb) {
    __shared__ unsigned short h_sh[H2_DIM];   // current hidden state, bf16
    __shared__ float          gw[G_DIM];      // h @ W_hh^T

    const int dir   = blockIdx.x;
    const int tid   = threadIdx.x;
    const int lane  = tid & 31;
    const int wv    = tid >> 5;               // 32 waves
    const int row16 = lane & 15;
    const int half  = lane >> 4;

    const float*          xw   = dir ? xw_b     : xw_f;
    const unsigned short* Whh  = dir ? Whh_b_bf : Whh_f_bf;
    float*                hout = dir ? hb       : hf;

    float c_reg = 0.0f;
    if (tid < H2_DIM) {
        c_reg = c0[dir * H2_DIM + tid];
        h_sh[tid] = f2bf(h0[dir * H2_DIM + tid]);
    }
    __syncthreads();

    for (int t = 0; t < L_SENT; ++t) {
        // ---- gate GEMV: each wave owns 4 gate tiles of 16 ----
#pragma unroll
        for (int nti = 0; nti < 4; ++nti) {
            const int nt = wv * 4 + nti;
            const int g  = nt * 16 + row16;
            const unsigned short* wBase = Whh + (size_t)g * H2_DIM;
            v8f acc = {};
            for (int k0 = 0; k0 < H2_DIM; k0 += 32) {
                BFFrag a, b;
                // broadcast h into every A row -> all D rows equal h.W
                a.u[0] = *(const u32x4*)(&h_sh[k0 + 8 * half]);
                a.u[1] = *(const u32x4*)(&h_sh[k0 + 16 + 8 * half]);
                b.u[0] = *(const u32x4*)(wBase + k0 + 8 * half);
                b.u[1] = *(const u32x4*)(wBase + k0 + 16 + 8 * half);
                acc = __builtin_amdgcn_wmma_f32_16x16x32_bf16(
                          false, a.v, false, b.v, (short)0, acc, false, false);
            }
            if (lane < 16) gw[nt * 16 + lane] = acc[0];  // row 0, col = lane
        }
        __syncthreads();

        // ---- gate nonlinearity + state update (threads 0..511) ----
        if (tid < H2_DIM) {
            const float* xr = xw + (size_t)t * G_DIM;
            if (t + 1 < L_SENT)
                __builtin_prefetch(xr + G_DIM + tid, 0, 0);
            float iv = sigmoidf_(xr[tid]                + gw[tid]);
            float fv = sigmoidf_(xr[H2_DIM + tid]       + gw[H2_DIM + tid]);
            float gv = tanhf(    xr[2 * H2_DIM + tid]   + gw[2 * H2_DIM + tid]);
            float ov = sigmoidf_(xr[3 * H2_DIM + tid]   + gw[3 * H2_DIM + tid]);
            c_reg = fv * c_reg + iv * gv;
            const float hval = ov * tanhf(c_reg);
            const int orow = dir ? (L_SENT - 1 - t) : t; // un-reverse bwd dir
            hout[(size_t)orow * H2_DIM + tid] = hval;
            h_sh[tid] = f2bf(hval);
        }
        __syncthreads();
    }
}

// ------------------------------------------------------------ output projection
__global__ void out_proj_kernel(const float* __restrict__ hf,
                                const float* __restrict__ hb,
                                const float* __restrict__ W_out,
                                const float* __restrict__ b_out,
                                float* __restrict__ feats) {
    int idx = blockIdx.x * blockDim.x + threadIdx.x;
    if (idx >= L_SENT * NTAG) return;
    const int t = idx / NTAG, tag = idx % NTAG;
    const float* wr  = W_out + (size_t)tag * (2 * H2_DIM);
    const float* hfr = hf + (size_t)t * H2_DIM;
    const float* hbr = hb + (size_t)t * H2_DIM;
    float s = b_out[tag];
    for (int j = 0; j < H2_DIM; ++j) s += hfr[j] * wr[j];
    for (int j = 0; j < H2_DIM; ++j) s += hbr[j] * wr[H2_DIM + j];
    feats[idx] = s;
}

// --------------------------------------------------------------- CRF forward
__global__ void crf_forward_kernel(const float* __restrict__ feats,
                                   const float* __restrict__ trans,
                                   float* __restrict__ out) {
    __shared__ float fv[NTAG];
    __shared__ float tr[NTAG * NTAG];
    const int tid = threadIdx.x;
    for (int i = tid; i < NTAG * NTAG; i += 32) tr[i] = trans[i];
    if (tid < NTAG) fv[tid] = (tid == START_TAG) ? 0.0f : NEG_INF;
    __syncthreads();

    for (int t = 0; t < L_SENT; ++t) {
        float nv = 0.0f;
        if (tid < NTAG) {
            const float feat = feats[t * NTAG + tid];
            float m = -3.4e38f;
            for (int j = 0; j < NTAG; ++j)
                m = fmaxf(m, fv[j] + tr[tid * NTAG + j]);
            float s = 0.0f;
            for (int j = 0; j < NTAG; ++j)
                s += __expf(fv[j] + tr[tid * NTAG + j] - m);
            nv = m + __logf(s) + feat;
        }
        __syncthreads();
        if (tid < NTAG) fv[tid] = nv;
        __syncthreads();
    }
    if (tid == 0) {
        float m = -3.4e38f;
        for (int j = 0; j < NTAG; ++j)
            m = fmaxf(m, fv[j] + tr[STOP_TAG * NTAG + j]);
        float s = 0.0f;
        for (int j = 0; j < NTAG; ++j)
            s += __expf(fv[j] + tr[STOP_TAG * NTAG + j] - m);
        out[0] = m + __logf(s);
    }
}

// ------------------------------------------------------------------- launcher
extern "C" void kernel_launch(void* const* d_in, const int* in_sizes, int n_in,
                              void* d_out, int out_size, void* d_ws, size_t ws_size,
                              hipStream_t stream) {
    const int*   sentence = (const int*)  d_in[0];
    const float* emb      = (const float*)d_in[1];
    const float* W_ih_f   = (const float*)d_in[2];
    const float* W_hh_f   = (const float*)d_in[3];
    const float* b_f      = (const float*)d_in[4];
    const float* W_ih_b   = (const float*)d_in[5];
    const float* W_hh_b   = (const float*)d_in[6];
    const float* b_b      = (const float*)d_in[7];
    const float* W_out    = (const float*)d_in[8];
    const float* b_out    = (const float*)d_in[9];
    const float* trans    = (const float*)d_in[10];
    const float* h0       = (const float*)d_in[11];
    const float* c0       = (const float*)d_in[12];

    char* ws = (char*)d_ws;
    const size_t MB = 1024ull * 1024ull;
    unsigned short* xs_bf    = (unsigned short*)(ws + 0);
    unsigned short* Wih_f_bf = (unsigned short*)(ws + 2 * MB);
    unsigned short* Wih_b_bf = (unsigned short*)(ws + 4 * MB);
    unsigned short* Whh_f_bf = (unsigned short*)(ws + 6 * MB);
    unsigned short* Whh_b_bf = (unsigned short*)(ws + 8 * MB);
    float* xw_f  = (float*)(ws + 10 * MB);
    float* xw_b  = (float*)(ws + 26 * MB);
    float* hf    = (float*)(ws + 42 * MB);
    float* hb    = (float*)(ws + 46 * MB);
    float* feats = (float*)(ws + 50 * MB);

    const int NW = G_DIM * H2_DIM;   // 2048*512 elements per weight matrix
    f32_to_bf16_kernel<<<NW / 256, 256, 0, stream>>>(W_ih_f, Wih_f_bf, NW);
    f32_to_bf16_kernel<<<NW / 256, 256, 0, stream>>>(W_ih_b, Wih_b_bf, NW);
    f32_to_bf16_kernel<<<NW / 256, 256, 0, stream>>>(W_hh_f, Whh_f_bf, NW);
    f32_to_bf16_kernel<<<NW / 256, 256, 0, stream>>>(W_hh_b, Whh_b_bf, NW);

    gather_embed_kernel<<<(L_SENT * E_DIM) / 256, 256, 0, stream>>>(
        sentence, emb, xs_bf);

    dim3 ggrid(L_SENT / 16, G_DIM / 16 / 8, 2);   // (128, 16, 2), 8 waves/block
    input_gemm_kernel<<<ggrid, 256, 0, stream>>>(
        xs_bf, Wih_f_bf, Wih_b_bf, b_f, b_b, xw_f, xw_b);

    lstm_recurrence_kernel<<<2, 1024, 0, stream>>>(
        xw_f, xw_b, Whh_f_bf, Whh_b_bf, h0, c0, hf, hb);

    out_proj_kernel<<<(L_SENT * NTAG + 255) / 256, 256, 0, stream>>>(
        hf, hb, W_out, b_out, feats);

    crf_forward_kernel<<<1, 32, 0, stream>>>(feats, trans, (float*)d_out);
}
